// DoubleMaskedChamferDistance_58308476010762
// MI455X (gfx1250) — compile-verified, matrix operand order checked by reading, not run on hardware
//
#include <hip/hip_runtime.h>
#include <hip/hip_bf16.h>

// ---------- types ----------
typedef __attribute__((ext_vector_type(16))) __bf16   v16bf;
typedef __attribute__((ext_vector_type(8)))  float    v8f;
typedef __attribute__((ext_vector_type(8)))  unsigned v8ui;
typedef __attribute__((ext_vector_type(4)))  unsigned v4ui;

// ---------- problem constants (from reference) ----------
constexpr int BB = 128;   // batch
constexpr int TV = 512;   // video positions
constexpr int TL = 64;    // lang positions
constexpr int DD = 512;   // feature dim
constexpr int KC = 128;   // K-chunk staged in LDS
constexpr int BSTR = 136; // padded LDS row stride (bf16 elems): 272B -> conflict-free

// ---------- split-bf16 helper ----------
// Truncation split: x = hi + lo + O(2^-16 x); hi = x & 0xFFFF0000, lo = bf16(x - hi).
// One v_perm_b32 packs two bf16 (high halves of two f32) into one dword.
__device__ __forceinline__ void split_pack(float x0, float x1,
                                           unsigned& hi, unsigned& lo, float& ss) {
  unsigned u0 = __float_as_uint(x0), u1 = __float_as_uint(x1);
  float h0 = __uint_as_float(u0 & 0xFFFF0000u);
  float h1 = __uint_as_float(u1 & 0xFFFF0000u);
  hi = __builtin_amdgcn_perm(u1, u0, 0x07060302u);
  lo = __builtin_amdgcn_perm(__float_as_uint(x1 - h1),
                             __float_as_uint(x0 - h0), 0x07060302u);
  ss = fmaf(x0, x0, ss);
  ss = fmaf(x1, x1, ss);
}

// ---------- init: zero the global-max cell ----------
__global__ void dmcd_init(unsigned* gmax) {
  if (threadIdx.x == 0 && blockIdx.x == 0) *gmax = 0u;
}

// ---------- pass 1: pd = ||a||^2 + ||b||^2 - 2*A*B^T via split-bf16 WMMA ----------
__global__ __launch_bounds__(256)
void dmcd_pd_wmma(const float* __restrict__ video,
                  const float* __restrict__ lang,
                  float* __restrict__ pd,
                  unsigned* __restrict__ gmax) {
  __shared__ __bf16 sBhi[TL][BSTR];
  __shared__ __bf16 sBlo[TL][BSTR];
  __shared__ float  sB2[TL];
  __shared__ float  sA2[8][16];

  const int b     = blockIdx.y;
  const int vtile = blockIdx.x;            // 0..3, 128 rows each
  const int tid   = threadIdx.x;
  const int wave  = tid >> 5;
  const int lane  = tid & 31;
  const int r     = lane & 15;             // row (A) / col (B) within 16
  const int h     = lane >> 4;             // half-wave selector
  const int vbase = vtile * 128 + wave * 16;

  if (tid < TL) sB2[tid] = 0.0f;

  const float* Arow = video + ((size_t)b * TV + (vbase + r)) * DD;
  const float* Lb   = lang + (size_t)b * TL * DD;

  v8f acc[4] = {};
  float a2part = 0.0f;

  for (int kc = 0; kc < DD; kc += KC) {
    __syncthreads();  // protect LDS reuse (and sB2 init on first pass)

    // ---- stage lang[b][0..63][kc..kc+KC) into LDS as bf16 hi/lo; accumulate ||b||^2
    {
      const int sr = tid & 63;             // lang row
      const int sq = tid >> 6;             // 0..3 -> 32 cols each
      const float4* s4 = (const float4*)(Lb + (size_t)sr * DD + kc + sq * 32);
      float ss = 0.0f;
      unsigned hp[16], lp[16];
#pragma unroll
      for (int q = 0; q < 8; ++q) {
        float4 c = s4[q];
        split_pack(c.x, c.y, hp[2 * q],     lp[2 * q],     ss);
        split_pack(c.z, c.w, hp[2 * q + 1], lp[2 * q + 1], ss);
      }
      uint4* dsth = (uint4*)&sBhi[sr][sq * 32];
      uint4* dstl = (uint4*)&sBlo[sr][sq * 32];
#pragma unroll
      for (int q = 0; q < 4; ++q) {
        dsth[q] = make_uint4(hp[4 * q], hp[4 * q + 1], hp[4 * q + 2], hp[4 * q + 3]);
        dstl[q] = make_uint4(lp[4 * q], lp[4 * q + 1], lp[4 * q + 2], lp[4 * q + 3]);
      }
      atomicAdd(&sB2[sr], ss);
    }
    __syncthreads();

    // prefetch next A chunk while we compute on this one
    if (kc + KC < DD) __builtin_prefetch(Arow + kc + KC + h * 8, 0, 0);

    // ---- 4 WMMA k-steps of 32 over this chunk
    for (int ks = 0; ks < KC; ks += 32) {
      const int k0 = kc + ks;
      // A fragment: lane holds row r, K = h*8..h*8+7 and 16+h*8..16+h*8+7
      const float* pa = Arow + k0 + h * 8;
      float4 cs[4];
      cs[0] = *(const float4*)(pa);
      cs[1] = *(const float4*)(pa + 4);
      cs[2] = *(const float4*)(pa + 16);
      cs[3] = *(const float4*)(pa + 20);
      v8ui hp, lp;
#pragma unroll
      for (int q = 0; q < 4; ++q) {
        unsigned h0, h1, l0, l1;
        split_pack(cs[q].x, cs[q].y, h0, l0, a2part);
        split_pack(cs[q].z, cs[q].w, h1, l1, a2part);
        hp[2 * q] = h0; hp[2 * q + 1] = h1;
        lp[2 * q] = l0; lp[2 * q + 1] = l1;
      }
      v16bf ahi = __builtin_bit_cast(v16bf, hp);
      v16bf alo = __builtin_bit_cast(v16bf, lp);
#pragma unroll
      for (int nt = 0; nt < 4; ++nt) {
        const int n   = nt * 16 + r;       // B column = lang row
        const int ksl = ks + h * 8;
        v4ui bh0 = *(const v4ui*)&sBhi[n][ksl];
        v4ui bh1 = *(const v4ui*)&sBhi[n][ksl + 16];
        v4ui bl0 = *(const v4ui*)&sBlo[n][ksl];
        v4ui bl1 = *(const v4ui*)&sBlo[n][ksl + 16];
        v8ui bh, bl;
#pragma unroll
        for (int i = 0; i < 4; ++i) {
          bh[i] = bh0[i]; bh[i + 4] = bh1[i];
          bl[i] = bl0[i]; bl[i + 4] = bl1[i];
        }
        v16bf bhi = __builtin_bit_cast(v16bf, bh);
        v16bf blo = __builtin_bit_cast(v16bf, bl);
        acc[nt] = __builtin_amdgcn_wmma_f32_16x16x32_bf16(
            false, ahi, false, bhi, (short)0, acc[nt], false, false);
        acc[nt] = __builtin_amdgcn_wmma_f32_16x16x32_bf16(
            false, alo, false, bhi, (short)0, acc[nt], false, false);
        acc[nt] = __builtin_amdgcn_wmma_f32_16x16x32_bf16(
            false, ahi, false, blo, (short)0, acc[nt], false, false);
      }
    }
  }

  // ---- row norms: combine the two K-halves held by lanes r and r+16
  float a2full = a2part + __shfl_xor(a2part, 16);
  if (h == 0) sA2[wave][r] = a2full;
  __syncthreads();

  // ---- epilogue: pd = a2 + b2 - 2S ; store tile; track max
  float tmax = 0.0f;
#pragma unroll
  for (int nt = 0; nt < 4; ++nt) {
    const int l  = nt * 16 + r;
    float b2 = sB2[l];
#pragma unroll
    for (int j = 0; j < 8; ++j) {
      const int vloc = h * 8 + j;          // C layout: VGPR j, halves by h
      float a2 = sA2[wave][vloc];
      float p  = a2 + b2 - 2.0f * acc[nt][j];
      pd[((size_t)b * TV + vbase + vloc) * TL + l] = p;
      tmax = fmaxf(tmax, p);
    }
  }
#pragma unroll
  for (int off = 16; off > 0; off >>= 1)
    tmax = fmaxf(tmax, __shfl_xor(tmax, off));
  if (lane == 0) atomicMax(gmax, __float_as_uint(fmaxf(tmax, 0.0f)));
}

// ---------- pass 2: masked chamfer reductions ----------
__device__ __forceinline__ float block_sum(float x, float* sred, int tid) {
  sred[tid] = x;
  __syncthreads();
  for (int s = 128; s > 0; s >>= 1) {
    if (tid < s) sred[tid] += sred[tid + s];
    __syncthreads();
  }
  float rv = sred[0];
  __syncthreads();
  return rv;
}

__global__ __launch_bounds__(256)
void dmcd_reduce(const float* __restrict__ pd,
                 const float* __restrict__ mask_v,
                 const float* __restrict__ mask_l,
                 const unsigned* __restrict__ gmax,
                 float* __restrict__ out) {
  __shared__ float sml[TL];
  __shared__ float smv[TV];
  __shared__ float sred[256];
  __shared__ float scol[4][TL];

  const int b = blockIdx.x;
  const int tid = threadIdx.x;
  const float mx = __uint_as_float(*gmax);
  const float* pdb = pd + (size_t)b * TV * TL;

  if (tid < TL) sml[tid] = mask_l[b * TL + tid];
  smv[tid]       = mask_v[b * TV + tid];
  smv[tid + 256] = mask_v[b * TV + tid + 256];
  __syncthreads();

  // rows: min over l, weighted by mask_v
  float accv = 0.0f, accNv = 0.0f;
  for (int v = tid; v < TV; v += 256) {
    float mvv = smv[v];
    accNv += mvv;
    float rmin = 3.4e38f;
    const float* row = pdb + (size_t)v * TL;
#pragma unroll 8
    for (int l = 0; l < TL; ++l)
      rmin = fminf(rmin, row[l] + (1.0f - mvv * sml[l]) * mx);
    accv += rmin * mvv;
  }
  float sumV = block_sum(accv,  sred, tid);
  float Nv   = block_sum(accNv, sred, tid);

  // cols: min over v, weighted by mask_l (4 segments of 128 rows per column)
  const int l   = tid & 63;
  const int seg = tid >> 6;
  float mll = sml[l];
  float cmin = 3.4e38f;
  for (int v = seg * 128; v < seg * 128 + 128; ++v)
    cmin = fminf(cmin, pdb[(size_t)v * TL + l] + (1.0f - smv[v] * mll) * mx);
  scol[seg][l] = cmin;
  __syncthreads();

  float partL = 0.0f, partNl = 0.0f;
  if (tid < TL) {
    float c = fminf(fminf(scol[0][tid], scol[1][tid]),
                    fminf(scol[2][tid], scol[3][tid]));
    partL  = c * sml[tid];
    partNl = sml[tid];
  }
  float sumL = block_sum(partL,  sred, tid);
  float Nl   = block_sum(partNl, sred, tid);

  if (tid == 0) out[b] = sumL / Nl + sumV / Nv;
}

extern "C" void kernel_launch(void* const* d_in, const int* in_sizes, int n_in,
                              void* d_out, int out_size, void* d_ws, size_t ws_size,
                              hipStream_t stream) {
  const float* video  = (const float*)d_in[0];
  const float* lang   = (const float*)d_in[1];
  const float* mask_v = (const float*)d_in[2];
  const float* mask_l = (const float*)d_in[3];
  float* out = (float*)d_out;

  unsigned* gmax = (unsigned*)d_ws;
  float* pd = (float*)((char*)d_ws + 256);   // 128*512*64 f32 = 16.8 MB scratch

  dmcd_init<<<1, 64, 0, stream>>>(gmax);
  dmcd_pd_wmma<<<dim3(TV / 128, BB), 256, 0, stream>>>(video, lang, pd, gmax);
  dmcd_reduce<<<BB, 256, 0, stream>>>(pd, mask_v, mask_l, gmax, out);
}
